// Encoder_15590731285097
// MI455X (gfx1250) — matrix operand order repaired and checked by reading, not compiled, not run on hardware
//
#include <hip/hip_runtime.h>
#include <math.h>

#define B_   8
#define N_   4096
#define CIN  64
#define CMID 128
#define COUT 64

typedef __attribute__((ext_vector_type(8)))  float        v8f;
typedef __attribute__((ext_vector_type(16))) __bf16       v16bf;
typedef __attribute__((ext_vector_type(4)))  unsigned int v4u;

union Frag16 { v16bf v; v4u u[2]; };

static __device__ __forceinline__ v8f wmma_bf16(v16bf a, v16bf b, v8f c) {
    return __builtin_amdgcn_wmma_f32_16x16x32_bf16(false, a, false, b,
                                                   (short)0, c, false, false);
}

// ---------------------------------------------------------------- sum of squares (pass 1)
__global__ __launch_bounds__(256) void k_sumsq(const float* __restrict__ x,
                                               float* __restrict__ partial) {
    __shared__ float red[256];
    int tid = threadIdx.x;
    size_t base = (size_t)blockIdx.x * 2048 + tid;   // 1024 blocks * 2048 = 2,097,152
    float s = 0.f;
#pragma unroll
    for (int i = 0; i < 8; i++) { float v = x[base + (size_t)i * 256]; s += v * v; }
    red[tid] = s; __syncthreads();
    for (int off = 128; off > 0; off >>= 1) {
        if (tid < off) red[tid] += red[tid + off];
        __syncthreads();
    }
    if (tid == 0) partial[blockIdx.x] = red[0];
}

// ---------------------------------------------------------------- scale (pass 2): inv_scale = sumsq^-1/4
__global__ __launch_bounds__(256) void k_scale(const float* __restrict__ partial,
                                               float* __restrict__ scl) {
    __shared__ float red[256];
    int tid = threadIdx.x;
    float s = partial[tid] + partial[tid + 256] + partial[tid + 512] + partial[tid + 768];
    red[tid] = s; __syncthreads();
    for (int off = 128; off > 0; off >>= 1) {
        if (tid < off) red[tid] += red[tid + off];
        __syncthreads();
    }
    if (tid == 0) scl[0] = 1.0f / sqrtf(sqrtf(red[0]));   // (sum x^2)^(-1/4)
}

// ---------------------------------------------------------------- q,k projection -> bf16
__global__ __launch_bounds__(128) void k_qk(const float* __restrict__ x,
                                            const float* __restrict__ Wq,
                                            const float* __restrict__ Wk,
                                            __bf16* __restrict__ qb,
                                            __bf16* __restrict__ kb) {
    __shared__ float xr[CIN];
    int bn = blockIdx.x;                 // (b*N + n)
    int tid = threadIdx.x;               // m = 0..127
    const float* xrow = x + (size_t)bn * CIN;
    if (tid < CIN) xr[tid] = xrow[tid];
    __syncthreads();
    float aq = 0.f, ak = 0.f;
#pragma unroll
    for (int c = 0; c < CIN; c++) {
        float xv = xr[c];
        aq += xv * Wq[c * CMID + tid];
        ak += xv * Wk[c * CMID + tid];
    }
    qb[(size_t)bn * CMID + tid] = (__bf16)aq;
    kb[(size_t)bn * CMID + tid] = (__bf16)ak;
}

// ---------------------------------------------------------------- x transpose -> bf16 hi/lo  [b][c][n]
__global__ __launch_bounds__(256) void k_xt(const float* __restrict__ x,
                                            __bf16* __restrict__ xh,
                                            __bf16* __restrict__ xl) {
    size_t i = (size_t)blockIdx.x * 256 + threadIdx.x;   // index into [b][c][n]
    int n = (int)(i & 4095);
    size_t bc = i >> 12;
    int c = (int)(bc & 63);
    size_t b = bc >> 6;
    float v = x[(b * N_ + n) * CIN + c];
    __bf16 h = (__bf16)v;
    __bf16 l = (__bf16)(v - (float)h);
    xh[i] = h; xl[i] = l;
}

// ---------------------------------------------------------------- fused attention + epilogue
// grid = N/16 workgroups; 8 waves per workgroup, wave b handles batch b.
__global__ __launch_bounds__(256) void k_attn(const __bf16* __restrict__ qb,
                                              const __bf16* __restrict__ kb,
                                              const __bf16* __restrict__ xth,
                                              const __bf16* __restrict__ xtl,
                                              const float* __restrict__ scl,
                                              const float* __restrict__ Wl,
                                              const float* __restrict__ bl,
                                              const float* __restrict__ g0,
                                              const float* __restrict__ b0,
                                              const float* __restrict__ g1,
                                              const float* __restrict__ b1,
                                              float* __restrict__ out) {
    // LDS carving:
    //  loop phase    : ebuf f32 [8][512] (16K) | pbuf_hi bf16 [8][512] (8K) | pbuf_lo (8K)
    //  epilogue phase: vbuf f32 [8][16][64] (32K, overlaps loop bufs)       | wl f32[4096] (16K)
    __shared__ __align__(16) char smem[49152];
    float*  ebuf    = (float*)smem;                     // [8][512]
    __bf16* pbuf_hi = (__bf16*)(smem + 16384);          // [8][512]
    __bf16* pbuf_lo = (__bf16*)(smem + 24576);          // [8][512]
    float*  vbuf    = (float*)smem;                     // [8][16][64] (epilogue)
    float*  wl      = (float*)(smem + 32768);           // [64][64]

    const int tid   = threadIdx.x;
    const int b     = tid >> 5;          // wave id == batch
    const int lane  = tid & 31;
    const int lhalf = lane >> 4;         // 0 | 1
    const int lm    = lane & 15;
    const int n0    = blockIdx.x * 16;
    const float inv_scale = scl[0];

    // resident A-frags: q rows n0..n0+15, K=0..127 (4 chunks of 32)
    Frag16 qf[4];
    {
        const __bf16* qrow = qb + ((size_t)b * N_ + n0 + lm) * CMID;
        int koff = lhalf ? 16 : 0;
#pragma unroll
        for (int kc = 0; kc < 4; kc++) {
            qf[kc].u[0] = *(const v4u*)(qrow + kc * 32 + koff);
            qf[kc].u[1] = *(const v4u*)(qrow + kc * 32 + 8 + koff);
        }
    }

    v8f acc[4];
#pragma unroll
    for (int i = 0; i < 4; i++)
#pragma unroll
        for (int j = 0; j < 8; j++) acc[i][j] = 0.f;

    for (int kt = 0; kt < N_ / 32; kt++) {
        const int nk0 = kt * 32;

        // ---- scores: s[16 x 32] = q_tile @ k_tile^T  (8 bf16 WMMAs)
        v8f s[2];
#pragma unroll
        for (int t = 0; t < 2; t++) {
#pragma unroll
            for (int j = 0; j < 8; j++) s[t][j] = 0.f;
            const __bf16* krow = kb + ((size_t)b * N_ + nk0 + t * 16 + lm) * CMID + lhalf * 16;
#pragma unroll
            for (int kc = 0; kc < 4; kc++) {
                Frag16 bf;
                bf.u[0] = *(const v4u*)(krow + kc * 32);
                bf.u[1] = *(const v4u*)(krow + kc * 32 + 8);
                s[t] = wmma_bf16(qf[kc].v, bf.v, s[t]);
            }
        }

        // ---- e = exp(s * inv_scale) -> LDS.
        // |s*inv_scale| << 1, so: no max-subtraction needed, and hardware
        // v_exp_f32 (__expf: 1 mul + 1 TRANS op, co-executes with WMMA) is
        // numerically equivalent to libm expf here (~1e-7 rel).
#pragma unroll
        for (int t = 0; t < 2; t++)
#pragma unroll
            for (int r = 0; r < 8; r++) {
                int M = r + lhalf * 8;
                int pos = M * 32 + t * 16 + lm;
                ebuf[b * 512 + pos] = __expf(s[t][r] * inv_scale);
            }
        __syncthreads();

        // ---- softmax over batch: each thread owns 2 adjacent positions
        {
            float e0[8], e1[8], s0 = 0.f, s1 = 0.f;
#pragma unroll
            for (int bb = 0; bb < 8; bb++) {
                float2 ev = *(const float2*)&ebuf[bb * 512 + tid * 2];
                e0[bb] = ev.x; e1[bb] = ev.y; s0 += ev.x; s1 += ev.y;
            }
            float i0 = 1.0f / s0, i1 = 1.0f / s1;
#pragma unroll
            for (int bb = 0; bb < 8; bb++) {
                float p0 = e0[bb] * i0, p1 = e1[bb] * i1;
                __bf16 h0 = (__bf16)p0, h1 = (__bf16)p1;
                __bf16 l0 = (__bf16)(p0 - (float)h0), l1 = (__bf16)(p1 - (float)h1);
                union { __bf16 h[2]; unsigned u; } ph, pl;
                ph.h[0] = h0; ph.h[1] = h1;
                pl.h[0] = l0; pl.h[1] = l1;
                *(unsigned*)&pbuf_hi[bb * 512 + tid * 2] = ph.u;
                *(unsigned*)&pbuf_lo[bb * 512 + tid * 2] = pl.u;
            }
        }
        __syncthreads();

        // ---- PV: acc[16 x 64] += p[16 x 32] @ x[32 x 64], split-precision (12 WMMAs)
        {
            int koff = lhalf ? 16 : 0;
            Frag16 pah, pal;
            pah.u[0] = *(const v4u*)&pbuf_hi[b * 512 + lm * 32 + koff];
            pah.u[1] = *(const v4u*)&pbuf_hi[b * 512 + lm * 32 + 8 + koff];
            pal.u[0] = *(const v4u*)&pbuf_lo[b * 512 + lm * 32 + koff];
            pal.u[1] = *(const v4u*)&pbuf_lo[b * 512 + lm * 32 + 8 + koff];
#pragma unroll
            for (int ct = 0; ct < 4; ct++) {
                const size_t xoff = ((size_t)b * CIN + ct * 16 + lm) * N_ + nk0 + lhalf * 16;
                Frag16 bh, blo;
                bh.u[0]  = *(const v4u*)(xth + xoff);
                bh.u[1]  = *(const v4u*)(xth + xoff + 8);
                blo.u[0] = *(const v4u*)(xtl + xoff);
                blo.u[1] = *(const v4u*)(xtl + xoff + 8);
                acc[ct] = wmma_bf16(pah.v, bh.v,  acc[ct]);
                acc[ct] = wmma_bf16(pah.v, blo.v, acc[ct]);
                acc[ct] = wmma_bf16(pal.v, bh.v,  acc[ct]);
            }
        }
        __syncthreads();   // ebuf reused next iteration
    }

    // ---- stage v_att into LDS, load Wl into LDS
#pragma unroll
    for (int ct = 0; ct < 4; ct++)
#pragma unroll
        for (int r = 0; r < 8; r++) {
            int M = r + lhalf * 8;
            vbuf[(b * 16 + M) * 64 + ct * 16 + lm] = acc[ct][r];
        }
#pragma unroll
    for (int i = 0; i < 16; i++) wl[tid + i * 256] = Wl[tid + i * 256];
    __syncthreads();

    // ---- epilogue: LN -> Wl + bl -> exact GELU -> LN. One thread per (batch,row).
    if (tid < 128) {
        int br = tid >> 4, M = tid & 15;
        const float* vr = vbuf + (br * 16 + M) * 64;
        float vals[64];
        float mu = 0.f;
#pragma unroll
        for (int c = 0; c < 64; c += 4) {
            float4 v4 = *(const float4*)(vr + c);
            vals[c] = v4.x; vals[c+1] = v4.y; vals[c+2] = v4.z; vals[c+3] = v4.w;
            mu += v4.x + v4.y + v4.z + v4.w;
        }
        mu *= (1.0f / 64.0f);
        float var = 0.f;
#pragma unroll
        for (int c = 0; c < 64; c++) { float d = vals[c] - mu; var += d * d; }
        var *= (1.0f / 64.0f);
        float rs = rsqrtf(var + 1e-5f);
#pragma unroll
        for (int c = 0; c < 64; c++)
            vals[c] = (vals[c] - mu) * rs * g0[c] + b0[c];

        float y[64];
        float mu2 = 0.f;
        for (int o = 0; o < 64; o++) {
            const float* wr = wl + o * 64;
            float a = bl[o];
#pragma unroll
            for (int c = 0; c < 64; c += 4) {
                float4 w4 = *(const float4*)(wr + c);
                a += vals[c] * w4.x + vals[c+1] * w4.y + vals[c+2] * w4.z + vals[c+3] * w4.w;
            }
            float g = 0.5f * a * (1.0f + erff(a * 0.70710678118654752f));
            y[o] = g;
            mu2 += g;
        }
        mu2 *= (1.0f / 64.0f);
        float var2 = 0.f;
#pragma unroll
        for (int o = 0; o < 64; o++) { float d = y[o] - mu2; var2 += d * d; }
        var2 *= (1.0f / 64.0f);
        float rs2 = rsqrtf(var2 + 1e-5f);

        float* orow = out + ((size_t)br * N_ + n0 + M) * 64;
#pragma unroll
        for (int o = 0; o < 64; o += 4) {
            float4 o4;
            o4.x = (y[o]   - mu2) * rs2 * g1[o]   + b1[o];
            o4.y = (y[o+1] - mu2) * rs2 * g1[o+1] + b1[o+1];
            o4.z = (y[o+2] - mu2) * rs2 * g1[o+2] + b1[o+2];
            o4.w = (y[o+3] - mu2) * rs2 * g1[o+3] + b1[o+3];
            *(float4*)(orow + o) = o4;
        }
    }
}

// ---------------------------------------------------------------- host launcher
extern "C" void kernel_launch(void* const* d_in, const int* in_sizes, int n_in,
                              void* d_out, int out_size, void* d_ws, size_t ws_size,
                              hipStream_t stream) {
    const float* x  = (const float*)d_in[0];
    const float* Wq = (const float*)d_in[1];
    const float* Wk = (const float*)d_in[2];
    const float* Wl = (const float*)d_in[3];
    const float* bl = (const float*)d_in[4];
    const float* g0 = (const float*)d_in[5];
    const float* b0 = (const float*)d_in[6];
    const float* g1 = (const float*)d_in[7];
    const float* b1 = (const float*)d_in[8];
    float* out = (float*)d_out;

    char* ws = (char*)d_ws;
    float*  partial = (float*)ws;                          // 4 KB
    float*  scl     = (float*)(ws + 4096);                 // 4 B
    __bf16* qb      = (__bf16*)(ws + 8192);                // 8 MB
    __bf16* kb      = qb + (size_t)B_ * N_ * CMID;         // 8 MB
    __bf16* xh      = kb + (size_t)B_ * N_ * CMID;         // 4 MB
    __bf16* xl      = xh + (size_t)B_ * CIN * N_;          // 4 MB

    k_sumsq<<<1024, 256, 0, stream>>>(x, partial);
    k_scale<<<1, 256, 0, stream>>>(partial, scl);
    k_qk<<<B_ * N_, 128, 0, stream>>>(x, Wq, Wk, qb, kb);
    k_xt<<<(B_ * N_ * CIN) / 256, 256, 0, stream>>>(x, xh, xl);
    k_attn<<<N_ / 16, 256, 0, stream>>>(qb, kb, xh, xl, scl,
                                        Wl, bl, g0, b0, g1, b1, out);
}
